// ALIGNN_50173807952901
// MI455X (gfx1250) — compile-verified, matrix operand order checked by reading, not run on hardware
//
#include <hip/hip_runtime.h>
#include <hip/hip_bf16.h>
#include <math.h>

// ---------------- problem constants (match reference) ----------------
#define HID   256
#define EMBD  64
#define AIF   92
#define EBINS 80
#define TBINS 40
#define NNODES 20000
#define NEDGES 120000
#define NTRIP  240000
#define NGRAPH 64

typedef __attribute__((ext_vector_type(16))) __bf16 v16bf;
typedef __attribute__((ext_vector_type(8)))  __bf16 v8bf;
typedef __attribute__((ext_vector_type(4)))  __bf16 v4bf;
typedef __attribute__((ext_vector_type(8)))  float  v8f;

__device__ __forceinline__ void atomAddF(float* p, float v) {
  __hip_atomic_fetch_add(p, v, __ATOMIC_RELAXED, __HIP_MEMORY_SCOPE_AGENT);
}

#define GS_LOOP(i, total)                                                     \
  for (size_t i = (size_t)blockIdx.x * blockDim.x + threadIdx.x; i < (total); \
       i += (size_t)gridDim.x * blockDim.x)

// ============================ kernels ============================

__global__ __launch_bounds__(256) void k_zero(float* __restrict__ p, size_t n) {
  GS_LOOP(i, n) p[i] = 0.0f;
}

__global__ __launch_bounds__(256) void k_rbf_angle(const float* __restrict__ a,
                                                   float* __restrict__ out) {
  const float gamma = (39.0f / 2.0f) * (39.0f / 2.0f);
  GS_LOOP(i, (size_t)NTRIP * TBINS) {
    int e = (int)(i / TBINS), bin = (int)(i % TBINS);
    float c  = -1.0f + 2.0f * (float)bin / 39.0f;
    float dd = a[e] - c;
    out[i] = __expf(-gamma * dd * dd);
  }
}

__global__ __launch_bounds__(256) void k_rbf_edge(const float* __restrict__ r,
                                                  float* __restrict__ out) {
  const float gamma = (79.0f / 8.0f) * (79.0f / 8.0f);
  GS_LOOP(i, (size_t)NEDGES * EBINS) {
    int e = (int)(i / EBINS), bin = (int)(i % EBINS);
    float x = r[3 * e], y = r[3 * e + 1], z = r[3 * e + 2];
    float d  = sqrtf(x * x + y * y + z * z);
    float c  = 8.0f * (float)bin / 79.0f;
    float dd = d - c;
    out[i] = __expf(-gamma * dd * dd);
  }
}

// -------- bf16 WMMA GEMM: C[M,N] = A[M,K] @ W[K,N] + bias --------
// 256 threads = 8 wave32 in 4(M) x 2(N); wave tile 32x64 = 2x4 WMMA tiles;
// block tile 128x128; K-step 32; double-buffered LDS (1 barrier / K-step,
// global loads of tile t+1 overlap WMMA of tile t). fp32 in/out, bf16
// compute, f32 accumulate. M/N tails via index clamping (feeds only
// discarded C entries). The K-tail tile is peeled OUT of the main loop so
// the steady state has a single, guard-free loader.
#define BM 128
#define BN 128
#define BK 32
#define LDT 40  // LDS row stride (bf16) = 32 + 8 pad -> 80B rows

template <bool GUARD_K>
__device__ __forceinline__ void load_tiles(
    const float* __restrict__ A, const float* __restrict__ W,
    __bf16* __restrict__ As, __bf16* __restrict__ Bs,
    int tid, int rowBase, int colBase, int M, int K, int N, int k0) {
  float4 va[4];
  float  vb[4][4];
  // ---- stage A: 128 rows x 32 K, one float4 per thread x 4 groups ----
#pragma unroll
  for (int it = 0; it < 4; ++it) {
    int linear = it * 256 + tid;
    int row = linear >> 3;
    int kk  = (linear & 7) << 2;
    int gr  = rowBase + row;
    if (gr > M - 1) gr = M - 1;          // clamp: feeds only discarded rows
    const float* ap = A + (size_t)gr * K + k0 + kk;
    if (!GUARD_K) {
      va[it] = *(const float4*)ap;
      __builtin_prefetch(ap + BK, 0, 1);         // next K slice (speculative)
    } else {
      float4 v; v.x = v.y = v.z = v.w = 0.0f;
      if (k0 + kk < K) v = *(const float4*)ap;   // K % 4 == 0
      va[it] = v;
    }
  }
  // ---- stage B: 32 K x 128 cols of W; thread owns 4 consecutive K of one
  //      column per group -> packs into a single b64 LDS store ----
#pragma unroll
  for (int it = 0; it < 4; ++it) {
    int linear = it * 256 + tid;
    int n  = linear & 127;
    int kq = (linear >> 7) << 2;         // 0,4,...,28
    int ng = colBase + n;
    if (ng > N - 1) ng = N - 1;          // clamp: feeds only discarded cols
#pragma unroll
    for (int j = 0; j < 4; ++j) {
      float v = 0.0f;
      if (!GUARD_K || (k0 + kq < K))     // K % 4 == 0 => whole group in/out
        v = W[(size_t)(k0 + kq + j) * N + ng];
      vb[it][j] = v;
    }
  }
  // ---- convert + store to LDS (b64 stores) ----
#pragma unroll
  for (int it = 0; it < 4; ++it) {
    int linear = it * 256 + tid;
    int row = linear >> 3;
    int kk  = (linear & 7) << 2;
    v4bf pa;
    pa[0] = (__bf16)va[it].x; pa[1] = (__bf16)va[it].y;
    pa[2] = (__bf16)va[it].z; pa[3] = (__bf16)va[it].w;
    *(v4bf*)&As[row * LDT + kk] = pa;
  }
#pragma unroll
  for (int it = 0; it < 4; ++it) {
    int linear = it * 256 + tid;
    int n  = linear & 127;
    int kq = (linear >> 7) << 2;
    v4bf pb;
    pb[0] = (__bf16)vb[it][0]; pb[1] = (__bf16)vb[it][1];
    pb[2] = (__bf16)vb[it][2]; pb[3] = (__bf16)vb[it][3];
    *(v4bf*)&Bs[n * LDT + kq] = pb;
  }
}

__device__ __forceinline__ void mma_step(const __bf16* __restrict__ As,
                                         const __bf16* __restrict__ Bs,
                                         int wm, int wn, int lr, int khalf,
                                         v8f acc[2][4]) {
  v16bf afrag[2], bfrag[4];
#pragma unroll
  for (int tm = 0; tm < 2; ++tm) {
    const __bf16* ap = &As[(wm * 32 + tm * 16 + lr) * LDT];
    v8bf lo = *(const v8bf*)(ap + khalf);        // K: khalf..khalf+7
    v8bf hi = *(const v8bf*)(ap + khalf + 16);   // K: khalf+16..+23
    afrag[tm] = __builtin_shufflevector(lo, hi, 0, 1, 2, 3, 4, 5, 6, 7,
                                        8, 9, 10, 11, 12, 13, 14, 15);
  }
#pragma unroll
  for (int tn = 0; tn < 4; ++tn) {
    const __bf16* bp = &Bs[(wn * 64 + tn * 16 + lr) * LDT];
    v8bf lo = *(const v8bf*)(bp + khalf);
    v8bf hi = *(const v8bf*)(bp + khalf + 16);
    bfrag[tn] = __builtin_shufflevector(lo, hi, 0, 1, 2, 3, 4, 5, 6, 7,
                                        8, 9, 10, 11, 12, 13, 14, 15);
  }
#pragma unroll
  for (int tm = 0; tm < 2; ++tm)
#pragma unroll
    for (int tn = 0; tn < 4; ++tn)
      acc[tm][tn] = __builtin_amdgcn_wmma_f32_16x16x32_bf16(
          false, afrag[tm], false, bfrag[tn], (short)0, acc[tm][tn],
          false, false);
}

__global__ __launch_bounds__(256) void k_gemm(
    const float* __restrict__ A, const float* __restrict__ W,
    const float* __restrict__ bias, float* __restrict__ C,
    int M, int K, int N) {
  __shared__ __bf16 As[2][BM * LDT];
  __shared__ __bf16 Bs[2][BN * LDT];

  const int tid  = threadIdx.x;
  const int lane = tid & 31;
  const int wave = tid >> 5;
  const int wm = wave >> 1;          // 0..3
  const int wn = wave & 1;           // 0..1
  const int lr = lane & 15;          // row/col within a 16-wide tile
  const int khalf = (lane < 16) ? 0 : 8;

  const int rowBase = blockIdx.y * BM;
  const int colBase = blockIdx.x * BN;

  v8f acc[2][4] = {};

  const int nfull   = K / BK;        // guard-free tiles
  const bool hasTail = (K & (BK - 1)) != 0;
  int cur = 0;

  if (nfull > 0) {
    // prologue: tile 0, guard-free
    load_tiles<false>(A, W, As[0], Bs[0], tid, rowBase, colBase, M, K, N, 0);
    __syncthreads();
    // steady state: only the guard-free loader lives in this loop
    for (int t = 0; t < nfull; ++t) {
      int nxt = t + 1;
      if (nxt < nfull)  // overlap next-tile global loads with this WMMA
        load_tiles<false>(A, W, As[cur ^ 1], Bs[cur ^ 1], tid, rowBase,
                          colBase, M, K, N, nxt * BK);
      mma_step(As[cur], Bs[cur], wm, wn, lr, khalf, acc);
      __syncthreads();
      cur ^= 1;
    }
  }
  if (hasTail) {  // at most one guarded tile, outside the hot loop
    load_tiles<true>(A, W, As[cur], Bs[cur], tid, rowBase, colBase, M, K, N,
                     nfull * BK);
    __syncthreads();
    mma_step(As[cur], Bs[cur], wm, wn, lr, khalf, acc);
  }

  // ---- epilogue: bias add, bounds-guarded stores ----
#pragma unroll
  for (int tm = 0; tm < 2; ++tm) {
    int rbase = rowBase + wm * 32 + tm * 16 + ((lane < 16) ? 0 : 8);
#pragma unroll
    for (int tn = 0; tn < 4; ++tn) {
      int col = colBase + wn * 64 + tn * 16 + lr;
      if (col < N) {
        float bv = bias ? bias[col] : 0.0f;
#pragma unroll
        for (int rr = 0; rr < 8; ++rr) {
          int row = rbase + rr;
          if (row < M) C[(size_t)row * N + col] = acc[tm][tn][rr] + bv;
        }
      }
    }
  }
}

// -------- BatchNorm (training-mode batch stats) + SiLU --------
__global__ __launch_bounds__(256) void k_colstats(const float* __restrict__ X,
                                                  int M, int C,
                                                  float* __restrict__ st,
                                                  int rpb) {
  int c  = threadIdx.x;
  int r0 = blockIdx.x * rpb;
  int r1 = r0 + rpb; if (r1 > M) r1 = M;
  float s = 0.0f, s2 = 0.0f;
  for (int r = r0; r < r1; ++r) {
    float v = X[(size_t)r * C + c];
    s += v; s2 += v * v;
  }
  atomAddF(&st[c], s);
  atomAddF(&st[C + c], s2);
}

__global__ __launch_bounds__(256) void k_bn_finalize(float* __restrict__ st,
                                                     int C, float invM) {
  int c = threadIdx.x;
  float mu  = st[c] * invM;
  float var = st[C + c] * invM - mu * mu;
  st[2 * C + c] = mu;
  st[3 * C + c] = rsqrtf(var + 1e-5f);
}

__global__ __launch_bounds__(256) void k_bn_silu(
    const float* __restrict__ t, const float* __restrict__ st,
    const float* __restrict__ g, const float* __restrict__ b,
    const float* __restrict__ xres, float* __restrict__ out,
    size_t total, int C) {
  const float* mu = st + 2 * C;
  const float* rs = st + 3 * C;
  int cm = C - 1;  // C is a power of two (64 or 256)
  GS_LOOP(i, total) {
    int c = (int)i & cm;
    float v  = (t[i] - mu[c]) * rs[c] * g[c] + b[c];
    float si = v / (1.0f + __expf(-v));
    out[i] = (xres ? xres[i] : 0.0f) + si;
  }
}

// -------- EGGC graph plumbing --------
__global__ __launch_bounds__(256) void k_gather_add(
    float* __restrict__ m, const float* __restrict__ Gs,
    const float* __restrict__ Gd, const int* __restrict__ s,
    const int* __restrict__ d, int ne) {
  GS_LOOP(i, (size_t)ne * HID) {
    int e = (int)(i >> 8), c = (int)(i & 255);
    m[i] += Gs[(size_t)s[e] * HID + c] + Gd[(size_t)d[e] * HID + c];
  }
}

__global__ __launch_bounds__(256) void k_scatter(
    const float* __restrict__ m, const float* __restrict__ Bh,
    const int* __restrict__ s, const int* __restrict__ d,
    float* __restrict__ num, float* __restrict__ den, int ne) {
  GS_LOOP(i, (size_t)ne * HID) {
    int e = (int)(i >> 8), c = (int)(i & 255);
    float sig = 1.0f / (1.0f + __expf(-m[i]));
    size_t off = (size_t)d[e] * HID + c;
    atomAddF(&num[off], Bh[(size_t)s[e] * HID + c] * sig);
    atomAddF(&den[off], sig);
  }
}

__global__ __launch_bounds__(256) void k_combine(float* __restrict__ t,
                                                 const float* __restrict__ num,
                                                 const float* __restrict__ den,
                                                 size_t total) {
  GS_LOOP(i, total) t[i] += num[i] / (den[i] + 1e-6f);
}

// -------- readout --------
__global__ __launch_bounds__(256) void k_pool(const float* __restrict__ X,
                                              const int* __restrict__ gid,
                                              float* __restrict__ hsum,
                                              float* __restrict__ cnt) {
  GS_LOOP(i, (size_t)NNODES * HID) {
    int v = (int)(i >> 8), c = (int)(i & 255);
    int g = gid[v];
    atomAddF(&hsum[(size_t)g * HID + c], X[i]);
    if (c == 0) atomAddF(&cnt[g], 1.0f);
  }
}

__global__ __launch_bounds__(256) void k_fc(const float* __restrict__ hsum,
                                            const float* __restrict__ cnt,
                                            const float* __restrict__ w,
                                            const float* __restrict__ b,
                                            float* __restrict__ out) {
  __shared__ float red[HID];
  int g = blockIdx.x, c = threadIdx.x;
  float cc = fmaxf(cnt[g], 1.0f);
  red[c] = (hsum[(size_t)g * HID + c] / cc) * w[c];
  __syncthreads();
  for (int s = HID / 2; s > 0; s >>= 1) {
    if (c < s) red[c] += red[c + s];
    __syncthreads();
  }
  if (c == 0) out[g] = red[0] + b[0];
}

// ============================ host side ============================

static inline int gsBlocks(size_t total) {
  size_t b = (total + 255) / 256;
  return (int)(b < 4096 ? b : 4096);
}

static void launch_gemm(const float* A, const float* W, const float* bias,
                        float* C, int M, int K, int N, hipStream_t s) {
  dim3 grid((unsigned)((N + BN - 1) / BN), (unsigned)((M + BM - 1) / BM));
  k_gemm<<<grid, 256, 0, s>>>(A, W, bias, C, M, K, N);
}

// t <- silu(bn(t)) [+ xres], written to `out`; St = 4*C floats of scratch.
static void stats_act(float* t, const float* g, const float* b,
                      const float* xres, float* out, int M, int C, float* St,
                      hipStream_t s) {
  k_zero<<<1, 256, 0, s>>>(St, (size_t)(2 * C));
  const int rpb  = 512;
  const int grid = (M + rpb - 1) / rpb;
  k_colstats<<<grid, C, 0, s>>>(t, M, C, St, rpb);
  k_bn_finalize<<<1, C, 0, s>>>(St, C, 1.0f / (float)M);
  size_t total = (size_t)M * C;
  k_bn_silu<<<gsBlocks(total), 256, 0, s>>>(t, St, g, b, xres, out, total, C);
}

struct EggcP {  // leaf order: JAX pytree (sorted dict keys)
  const float *bne_b, *bne_g, *bnn_b, *bnn_g;
  const float *dg_b, *dg_w, *du_b, *du_w, *eg_b, *eg_w;
  const float *sg_b, *sg_w, *su_b, *su_w;
};
static EggcP eggcAt(void* const* d_in, int base) {
  EggcP p;
  p.bne_b = (const float*)d_in[base + 0];  p.bne_g = (const float*)d_in[base + 1];
  p.bnn_b = (const float*)d_in[base + 2];  p.bnn_g = (const float*)d_in[base + 3];
  p.dg_b  = (const float*)d_in[base + 4];  p.dg_w  = (const float*)d_in[base + 5];
  p.du_b  = (const float*)d_in[base + 6];  p.du_w  = (const float*)d_in[base + 7];
  p.eg_b  = (const float*)d_in[base + 8];  p.eg_w  = (const float*)d_in[base + 9];
  p.sg_b  = (const float*)d_in[base + 10]; p.sg_w  = (const float*)d_in[base + 11];
  p.su_b  = (const float*)d_in[base + 12]; p.su_w  = (const float*)d_in[base + 13];
  return p;
}

struct Ws {
  float *X, *Y, *Z, *Gs, *Gd, *Bh, *Tt, *Mb, *St, *Hsum, *Cnt;
};

static void run_eggc(const EggcP& p, const int* s, const int* d, float* x,
                     float* e, int nx, int ne, const Ws& w, hipStream_t st) {
  launch_gemm(x, p.sg_w, p.sg_b, w.Gs, nx, HID, HID, st);
  launch_gemm(x, p.dg_w, p.dg_b, w.Gd, nx, HID, HID, st);
  launch_gemm(e, p.eg_w, p.eg_b, w.Mb, ne, HID, HID, st);
  k_gather_add<<<gsBlocks((size_t)ne * HID), 256, 0, st>>>(w.Mb, w.Gs, w.Gd, s, d, ne);
  launch_gemm(x, p.du_w, p.du_b, w.Bh, nx, HID, HID, st);
  // reuse Gs/Gd as num/den
  k_zero<<<gsBlocks((size_t)nx * HID), 256, 0, st>>>(w.Gs, (size_t)nx * HID);
  k_zero<<<gsBlocks((size_t)nx * HID), 256, 0, st>>>(w.Gd, (size_t)nx * HID);
  k_scatter<<<gsBlocks((size_t)ne * HID), 256, 0, st>>>(w.Mb, w.Bh, s, d, w.Gs, w.Gd, ne);
  launch_gemm(x, p.su_w, p.su_b, w.Tt, nx, HID, HID, st);
  k_combine<<<gsBlocks((size_t)nx * HID), 256, 0, st>>>(w.Tt, w.Gs, w.Gd, (size_t)nx * HID);
  stats_act(w.Tt, p.bnn_g, p.bnn_b, /*xres=*/x, /*out=*/x, nx, HID, w.St, st);
  stats_act(w.Mb, p.bne_g, p.bne_b, /*xres=*/e, /*out=*/e, ne, HID, w.St, st);
}

extern "C" void kernel_launch(void* const* d_in, const int* in_sizes, int n_in,
                              void* d_out, int out_size, void* d_ws,
                              size_t ws_size, hipStream_t stream) {
  (void)in_sizes; (void)n_in; (void)out_size; (void)ws_size;
  // ---- input indices (top-level dict insertion order; nested params in
  //      JAX pytree order = sorted dict keys, lists in order) ----
  //  0..13  alignn[0]["edge"]   14..27 alignn[0]["node"]
  // 28..41  alignn[1]["edge"]   42..55 alignn[1]["node"]
  // 56..59  angle_emb1 (bn.b, bn.g, lin.b, lin.w)
  // 60..63  angle_emb2   64..67 atom_emb
  // 68..71  edge_emb1    72..75 edge_emb2
  // 76 fc.b   77 fc.w
  // 78..91  gcn[0]   92..105 gcn[1]
  // 106 atom_features  107 r  108 angle_h  109 src  110 dst
  // 111 lg_src  112 lg_dst  113 node_graph_ids  114 num_graphs
  EggcP al0e = eggcAt(d_in, 0),  al0n = eggcAt(d_in, 14);
  EggcP al1e = eggcAt(d_in, 28), al1n = eggcAt(d_in, 42);
  EggcP gcn0 = eggcAt(d_in, 78), gcn1 = eggcAt(d_in, 92);
  const float* ae1_bnb = (const float*)d_in[56];
  const float* ae1_bng = (const float*)d_in[57];
  const float* ae1_b   = (const float*)d_in[58];
  const float* ae1_w   = (const float*)d_in[59];
  const float* ae2_bnb = (const float*)d_in[60];
  const float* ae2_bng = (const float*)d_in[61];
  const float* ae2_b   = (const float*)d_in[62];
  const float* ae2_w   = (const float*)d_in[63];
  const float* at_bnb  = (const float*)d_in[64];
  const float* at_bng  = (const float*)d_in[65];
  const float* at_b    = (const float*)d_in[66];
  const float* at_w    = (const float*)d_in[67];
  const float* ee1_bnb = (const float*)d_in[68];
  const float* ee1_bng = (const float*)d_in[69];
  const float* ee1_b   = (const float*)d_in[70];
  const float* ee1_w   = (const float*)d_in[71];
  const float* ee2_bnb = (const float*)d_in[72];
  const float* ee2_bng = (const float*)d_in[73];
  const float* ee2_b   = (const float*)d_in[74];
  const float* ee2_w   = (const float*)d_in[75];
  const float* fc_b    = (const float*)d_in[76];
  const float* fc_w    = (const float*)d_in[77];
  const float* atomf   = (const float*)d_in[106];
  const float* rvec    = (const float*)d_in[107];
  const float* angh    = (const float*)d_in[108];
  const int* src    = (const int*)d_in[109];
  const int* dst    = (const int*)d_in[110];
  const int* lg_src = (const int*)d_in[111];
  const int* lg_dst = (const int*)d_in[112];
  const int* gid    = (const int*)d_in[113];
  float* out = (float*)d_out;

  // ---- workspace partition (floats); ~1.05 GiB total ----
  Ws w;
  float* base = (float*)d_ws;
  w.X  = base;                               // [N,256]
  w.Y  = w.X  + (size_t)NNODES * HID;        // [E,256]
  w.Z  = w.Y  + (size_t)NEDGES * HID;        // [T,256]
  w.Gs = w.Z  + (size_t)NTRIP  * HID;        // [E,256] (also num / rbf scratch)
  w.Gd = w.Gs + (size_t)NEDGES * HID;        // [E,256] (also den / emb scratch)
  w.Bh = w.Gd + (size_t)NEDGES * HID;        // [E,256]
  w.Tt = w.Bh + (size_t)NEDGES * HID;        // [E,256]
  w.Mb = w.Tt + (size_t)NEDGES * HID;        // [T,256]
  w.St = w.Mb + (size_t)NTRIP  * HID;        // 4*256 stats
  w.Hsum = w.St + 4 * HID;                   // [G,256]
  w.Cnt  = w.Hsum + (size_t)NGRAPH * HID;    // [G]

  hipStream_t s = stream;

  // ---- angle embedding: z = mlp2(mlp1(rbf(angle_h))) ----
  k_rbf_angle<<<gsBlocks((size_t)NTRIP * TBINS), 256, 0, s>>>(angh, w.Gs);
  launch_gemm(w.Gs, ae1_w, ae1_b, w.Gd, NTRIP, TBINS, EMBD, s);
  stats_act(w.Gd, ae1_bng, ae1_bnb, nullptr, w.Gd, NTRIP, EMBD, w.St, s);
  launch_gemm(w.Gd, ae2_w, ae2_b, w.Z, NTRIP, EMBD, HID, s);
  stats_act(w.Z, ae2_bng, ae2_bnb, nullptr, w.Z, NTRIP, HID, w.St, s);

  // ---- atom embedding: x = mlp(atom_features) ----
  launch_gemm(atomf, at_w, at_b, w.X, NNODES, AIF, HID, s);
  stats_act(w.X, at_bng, at_bnb, nullptr, w.X, NNODES, HID, w.St, s);

  // ---- edge embedding: y = mlp2(mlp1(rbf(|r|))) ----
  k_rbf_edge<<<gsBlocks((size_t)NEDGES * EBINS), 256, 0, s>>>(rvec, w.Gs);
  launch_gemm(w.Gs, ee1_w, ee1_b, w.Gd, NEDGES, EBINS, EMBD, s);
  stats_act(w.Gd, ee1_bng, ee1_bnb, nullptr, w.Gd, NEDGES, EMBD, w.St, s);
  launch_gemm(w.Gd, ee2_w, ee2_b, w.Y, NEDGES, EMBD, HID, s);
  stats_act(w.Y, ee2_bng, ee2_bnb, nullptr, w.Y, NEDGES, HID, w.St, s);

  // ---- ALIGNN layers: node conv (X,Y)->(X, m in Y); edge conv (Y,Z) ----
  run_eggc(al0n, src, dst, w.X, w.Y, NNODES, NEDGES, w, s);
  run_eggc(al0e, lg_src, lg_dst, w.Y, w.Z, NEDGES, NTRIP, w, s);
  run_eggc(al1n, src, dst, w.X, w.Y, NNODES, NEDGES, w, s);
  run_eggc(al1e, lg_src, lg_dst, w.Y, w.Z, NEDGES, NTRIP, w, s);

  // ---- GCN layers ----
  run_eggc(gcn0, src, dst, w.X, w.Y, NNODES, NEDGES, w, s);
  run_eggc(gcn1, src, dst, w.X, w.Y, NNODES, NEDGES, w, s);

  // ---- graph mean-pool + FC ----
  k_zero<<<gsBlocks((size_t)NGRAPH * HID + NGRAPH), 256, 0, s>>>(
      w.Hsum, (size_t)NGRAPH * HID + NGRAPH);
  k_pool<<<gsBlocks((size_t)NNODES * HID), 256, 0, s>>>(w.X, gid, w.Hsum, w.Cnt);
  k_fc<<<NGRAPH, HID, 0, s>>>(w.Hsum, w.Cnt, fc_w, fc_b, out);
}
